// MoE_57131654971569
// MI455X (gfx1250) — matrix operand order repaired and checked by reading, not compiled, not run on hardware
//
#include <hip/hip_runtime.h>

// ---------------- problem constants (match reference) ----------------
#define H      2880
#define E      8
#define FF     2880
#define TWOFF  5760
#define S      2048          // B*T = 2*1024
#define KDIM2  (E * FF)      // 23040
#define CLIP   7.0f
#define AUX_COEF 0.01f

// ---------------- types ----------------
typedef __bf16 bf16;
typedef __attribute__((ext_vector_type(16))) __bf16 v16bf;
typedef __attribute__((ext_vector_type(8)))  __bf16 v8bf;
typedef __attribute__((ext_vector_type(8)))  float  v8f;
typedef __attribute__((ext_vector_type(4)))  float  v4f;

#define APAD 40   // LDS row pitch in bf16 elements (32 data + 8 pad), keeps 16B alignment

__device__ __forceinline__ v16bf cat8(v8bf lo, v8bf hi) {
  return __builtin_shufflevector(lo, hi, 0,1,2,3,4,5,6,7,8,9,10,11,12,13,14,15);
}

// =====================================================================
// Router: one wave (32 lanes) per token. logits = x @ router_w.T + b.
// Writes final_w [S,E], probs [S,E], entropy [S] to workspace.
// =====================================================================
__global__ __launch_bounds__(256) void router_kernel(
    const float* __restrict__ x, const float* __restrict__ rw,
    const float* __restrict__ rb, float* __restrict__ final_w,
    float* __restrict__ probs_ws, float* __restrict__ ent_ws)
{
  const int wave = threadIdx.x >> 5;
  const int lane = threadIdx.x & 31;
  const int s    = blockIdx.x * 8 + wave;

  float acc[E];
#pragma unroll
  for (int e = 0; e < E; ++e) acc[e] = 0.f;

  const float* xr = x + (size_t)s * H;
  for (int h = lane; h < H; h += 32) {
    const float xv = xr[h];
#pragma unroll
    for (int e = 0; e < E; ++e) acc[e] += xv * rw[e * H + h];
  }
#pragma unroll
  for (int e = 0; e < E; ++e) {
#pragma unroll
    for (int off = 16; off > 0; off >>= 1)
      acc[e] += __shfl_xor(acc[e], off, 32);
  }

  if (lane == 0) {
    float lg[E];
#pragma unroll
    for (int e = 0; e < E; ++e) lg[e] = acc[e] + rb[e];

    // top-2 (first-index wins ties, matching lax.top_k)
    int i1 = 0;
#pragma unroll
    for (int e = 1; e < E; ++e) if (lg[e] > lg[i1]) i1 = e;
    int i2 = -1;
#pragma unroll
    for (int e = 0; e < E; ++e) {
      if (e == i1) continue;
      if (i2 < 0 || lg[e] > lg[i2]) i2 = e;
    }
    const float l1 = lg[i1], l2 = lg[i2];
    const float e2 = __expf(l2 - l1);
    const float w1 = 1.f / (1.f + e2);
    const float w2 = e2 / (1.f + e2);

    float fw[E];
#pragma unroll
    for (int e = 0; e < E; ++e) fw[e] = 0.f;
    fw[i1] = w1; fw[i2] = w2;

    float p[E], sum = 0.f;
#pragma unroll
    for (int e = 0; e < E; ++e) { p[e] = __expf(lg[e] - l1); sum += p[e]; }
    const float inv = 1.f / sum;
    float ent = 0.f;
#pragma unroll
    for (int e = 0; e < E; ++e) { p[e] *= inv; ent -= p[e] * __logf(p[e] + 1e-10f); }

#pragma unroll
    for (int e = 0; e < E; ++e) {
      final_w[s * E + e]  = fw[e];
      probs_ws[s * E + e] = p[e];
    }
    ent_ws[s] = ent;
  }
}

// =====================================================================
// GEMM1: expert_in = x @ W_in[e] (+b_in), clip, SiLU-gate, scale by
// final_w[s,e], write bf16 act[s, e*FF+f].
// WG: 256 thr = 8 waves (4M x 2N). WG tile M=256, N=64 (f-strip 32,
// up/gate interleaved per 16 cols). Wave tile 64x32 -> 8 accumulators.
// Double-buffered LDS, one barrier per K-step.
// =====================================================================
__global__ __launch_bounds__(256) void gemm1_kernel(
    const float* __restrict__ x, const float* __restrict__ W_in,
    const float* __restrict__ b_in, const float* __restrict__ final_w,
    bf16* __restrict__ act)
{
  __shared__ __align__(16) bf16 At[2][256 * APAD];
  __shared__ __align__(16) bf16 Bt[2][64 * APAD];

  const int tid   = threadIdx.x;
  const int wave  = tid >> 5;
  const int lane  = tid & 31;
  const int mwave = wave >> 1;          // 0..3
  const int nwave = wave & 1;           // 0..1
  const int mWG   = blockIdx.x * 256;
  const int fWG   = blockIdx.y * 32;
  const int e     = blockIdx.z;
  const float* Wb = W_in + (size_t)e * H * TWOFF;

  const v8f vzero = {0.f,0.f,0.f,0.f,0.f,0.f,0.f,0.f};
  v8f acc[8];
#pragma unroll
  for (int i = 0; i < 8; ++i) acc[i] = vzero;

  // ---- staging indices: A = one 32-float row per thread
  const float* arow_ptr = x + (size_t)(mWG + tid) * H;
  // B: 32x64, thread covers k=tid>>3, col chunk of 8.  Column interleave:
  // grp = c>>4: 0=up f[0:16) 1=gate f[0:16) 2=up f[16:32) 3=gate f[16:32)
  const int bk  = tid >> 3;
  const int bc  = (tid & 7) * 8;
  const int grp = bc >> 4;
  const int f0  = (grp >> 1) * 16 + (bc & 15);
  const int gcolbase = (grp & 1) ? (FF + fWG + f0) : (fWG + f0);

  // ---- fragment indices
  const int lc  = lane & 15;
  const int kha = (lane & 16) ? 8 : 0;
  const int khb = (lane & 16) ? 8 : 0;

  auto load_regs = [&](int kk, v4f a[8], v4f b[2]) {
    const float* ap = arow_ptr + kk;
#pragma unroll
    for (int i = 0; i < 8; ++i) a[i] = *(const v4f*)(ap + 4 * i);
    const float* bp = Wb + (size_t)(kk + bk) * TWOFF + gcolbase;
    b[0] = *(const v4f*)(bp + 0);
    b[1] = *(const v4f*)(bp + 4);
  };
  auto store_lds = [&](int buf, v4f a[8], v4f b[2]) {
    bf16* Arow = &At[buf][tid * APAD];
#pragma unroll
    for (int i = 0; i < 4; ++i) {
      v8bf p;
#pragma unroll
      for (int j = 0; j < 4; ++j) { p[j] = (bf16)a[2*i][j]; p[4+j] = (bf16)a[2*i+1][j]; }
      *(v8bf*)(Arow + 8 * i) = p;
    }
#pragma unroll
    for (int j = 0; j < 4; ++j) {
      Bt[buf][(bc + j)     * APAD + bk] = (bf16)b[0][j];
      Bt[buf][(bc + 4 + j) * APAD + bk] = (bf16)b[1][j];
    }
  };
  auto compute = [&](int buf) {
    v16bf bfrag[2];
#pragma unroll
    for (int n = 0; n < 2; ++n) {
      const int bn = nwave * 32 + n * 16 + lc;
      v8bf blo = *(const v8bf*)&Bt[buf][bn * APAD + khb];
      v8bf bhi = *(const v8bf*)&Bt[buf][bn * APAD + khb + 16];
      bfrag[n] = cat8(blo, bhi);
    }
#pragma unroll
    for (int m = 0; m < 4; ++m) {
      const int arow = mwave * 64 + m * 16 + lc;
      v8bf alo = *(const v8bf*)&At[buf][arow * APAD + kha];
      v8bf ahi = *(const v8bf*)&At[buf][arow * APAD + kha + 16];
      v16bf afrag = cat8(alo, ahi);
#pragma unroll
      for (int n = 0; n < 2; ++n)
        acc[m * 2 + n] = __builtin_amdgcn_wmma_f32_16x16x32_bf16(
            false, afrag, false, bfrag[n], (short)0, acc[m * 2 + n], false, false);
    }
  };

  // prologue
  { v4f a[8], b[2]; load_regs(0, a, b); store_lds(0, a, b); }
  __syncthreads();
  int buf = 0;
  for (int kk = 32; kk < H; kk += 32) {
    v4f a[8], b[2];
    load_regs(kk, a, b);     // issue global loads
    compute(buf);            // WMMAs cover the latency
    store_lds(buf ^ 1, a, b);
    __syncthreads();
    buf ^= 1;
  }
  compute(buf);

  // ---- epilogue: (up,gate) pair is (nsub0,nsub1) of this wave
  const int rb0 = mWG + mwave * 64 + ((lane & 16) ? 8 : 0);
  const int f   = fWG + nwave * 16 + lc;
  const float bu = b_in[e * TWOFF + f];
  const float bg = b_in[e * TWOFF + FF + f];
#pragma unroll
  for (int m = 0; m < 4; ++m) {
    v8f ua = acc[m * 2 + 0];
    v8f ga = acc[m * 2 + 1];
#pragma unroll
    for (int v = 0; v < 8; ++v) {
      const int r = rb0 + m * 16 + v;
      const float fw = final_w[r * E + e];
      float u = fminf(fmaxf(ua[v] + bu, -CLIP), CLIP);
      float g = fminf(fmaxf(ga[v] + bg, -CLIP), CLIP);
      act[(size_t)r * KDIM2 + e * FF + f] = (bf16)((g / (1.f + __expf(-g))) * u * fw);
    }
  }
}

// =====================================================================
// GEMM2: out = act_scaled @ W_out_flat (+ sum_e final_w*b_out)
// A tile (pure bf16 copy) staged with GLOBAL_LOAD_ASYNC_TO_LDS_B128
// (ASYNCcnt path); B tile f32->bf16 via VALU. Same 8-wave/8-acc scheme.
// =====================================================================
__global__ __launch_bounds__(256) void gemm2_kernel(
    const bf16* __restrict__ act, const float* __restrict__ W_out,
    const float* __restrict__ b_out, const float* __restrict__ final_w,
    float* __restrict__ out)
{
  __shared__ __align__(16) bf16 At[2][256 * APAD];
  __shared__ __align__(16) bf16 Bt[2][64 * APAD];

  const int tid   = threadIdx.x;
  const int wave  = tid >> 5;
  const int lane  = tid & 31;
  const int mwave = wave >> 1;
  const int nwave = wave & 1;
  const int mWG   = blockIdx.x * 256;
  const int nWG   = blockIdx.y * 64;

  const v8f vzero = {0.f,0.f,0.f,0.f,0.f,0.f,0.f,0.f};
  v8f acc[8];
#pragma unroll
  for (int i = 0; i < 8; ++i) acc[i] = vzero;

  // A: one 32-bf16 (64B) row segment per thread, async copied to LDS
  const bf16* arow_ptr = act + (size_t)(mWG + tid) * KDIM2;
  unsigned lds_a0 = (unsigned)(size_t)&At[0][tid * APAD];
  unsigned lds_a1 = (unsigned)(size_t)&At[1][tid * APAD];

  const int bk = tid >> 3;
  const int bc = (tid & 7) * 8;

  const int lc  = lane & 15;
  const int kha = (lane & 16) ? 8 : 0;
  const int khb = (lane & 16) ? 8 : 0;

  auto stage_A_async = [&](int buf, int kk) {
    unsigned long long ga = (unsigned long long)(size_t)(arow_ptr + kk);
    unsigned la = buf ? lds_a1 : lds_a0;
    // instruction offset applies to BOTH the LDS and global addresses
    asm volatile(
        "global_load_async_to_lds_b128 %0, %1, off\n\t"
        "global_load_async_to_lds_b128 %0, %1, off offset:16\n\t"
        "global_load_async_to_lds_b128 %0, %1, off offset:32\n\t"
        "global_load_async_to_lds_b128 %0, %1, off offset:48"
        :: "v"(la), "v"(ga) : "memory");
  };
  auto load_B = [&](int kk, v4f b[2]) {
    const float* bp = W_out + (size_t)(kk + bk) * H + nWG + bc;
    b[0] = *(const v4f*)(bp + 0);
    b[1] = *(const v4f*)(bp + 4);
  };
  auto store_B = [&](int buf, v4f b[2]) {
#pragma unroll
    for (int j = 0; j < 4; ++j) {
      Bt[buf][(bc + j)     * APAD + bk] = (bf16)b[0][j];
      Bt[buf][(bc + 4 + j) * APAD + bk] = (bf16)b[1][j];
    }
  };
  auto compute = [&](int buf) {
    v16bf bfrag[2];
#pragma unroll
    for (int n = 0; n < 2; ++n) {
      const int bn = nwave * 32 + n * 16 + lc;
      v8bf blo = *(const v8bf*)&Bt[buf][bn * APAD + khb];
      v8bf bhi = *(const v8bf*)&Bt[buf][bn * APAD + khb + 16];
      bfrag[n] = cat8(blo, bhi);
    }
#pragma unroll
    for (int m = 0; m < 4; ++m) {
      const int arow = mwave * 64 + m * 16 + lc;
      v8bf alo = *(const v8bf*)&At[buf][arow * APAD + kha];
      v8bf ahi = *(const v8bf*)&At[buf][arow * APAD + kha + 16];
      v16bf afrag = cat8(alo, ahi);
#pragma unroll
      for (int n = 0; n < 2; ++n)
        acc[m * 2 + n] = __builtin_amdgcn_wmma_f32_16x16x32_bf16(
            false, afrag, false, bfrag[n], (short)0, acc[m * 2 + n], false, false);
    }
  };

  // prologue
  stage_A_async(0, 0);
  { v4f b[2]; load_B(0, b); store_B(0, b); }
  asm volatile("s_wait_asynccnt 0x0" ::: "memory");
  __syncthreads();
  int buf = 0;
  for (int kk = 32; kk < KDIM2; kk += 32) {
    stage_A_async(buf ^ 1, kk);   // async: HW DMA to LDS, no VGPR round trip
    v4f b[2];
    load_B(kk, b);
    compute(buf);
    store_B(buf ^ 1, b);
    asm volatile("s_wait_asynccnt 0x0" ::: "memory");
    __syncthreads();
    buf ^= 1;
  }
  compute(buf);

  // ---- epilogue: add sum_e final_w[s,e]*b_out[e,h], store f32
  const int rb0 = mWG + mwave * 64 + ((lane & 16) ? 8 : 0);
  float bo[2][E];
#pragma unroll
  for (int n = 0; n < 2; ++n) {
    const int hcol = nWG + nwave * 32 + n * 16 + lc;
#pragma unroll
    for (int e = 0; e < E; ++e) bo[n][e] = b_out[e * H + hcol];
  }
#pragma unroll
  for (int m = 0; m < 4; ++m) {
#pragma unroll
    for (int v = 0; v < 8; ++v) {
      const int r = rb0 + m * 16 + v;
      const float* fwp = &final_w[r * E];
      float fwv[E];
#pragma unroll
      for (int e = 0; e < E; ++e) fwv[e] = fwp[e];
#pragma unroll
      for (int n = 0; n < 2; ++n) {
        float t = acc[m * 2 + n][v];
#pragma unroll
        for (int e = 0; e < E; ++e) t += fwv[e] * bo[n][e];
        out[(size_t)r * H + (nWG + nwave * 32 + n * 16 + lc)] = t;
      }
    }
  }
}

// =====================================================================
// Deterministic reduction of aux stats: one block, tree reduce.
// tail[0]=aux_loss, tail[1..8]=expert_utilization, tail[9]=entropy
// =====================================================================
__global__ __launch_bounds__(256) void reduce_stats_kernel(
    const float* __restrict__ final_w, const float* __restrict__ probs_ws,
    const float* __restrict__ ent_ws, float* __restrict__ tail)
{
  __shared__ float red[256];
  __shared__ float sums[17];
  const int t = threadIdx.x;

  float part[17];
#pragma unroll
  for (int q = 0; q < 17; ++q) part[q] = 0.f;

  for (int s = t; s < S; s += 256) {
#pragma unroll
    for (int e = 0; e < E; ++e) part[e]     += probs_ws[s * E + e];
#pragma unroll
    for (int e = 0; e < E; ++e) part[8 + e] += final_w[s * E + e];
    part[16] += ent_ws[s];
  }

  for (int q = 0; q < 17; ++q) {
    red[t] = part[q];
    __syncthreads();
    for (int off = 128; off > 0; off >>= 1) {
      if (t < off) red[t] += red[t + off];
      __syncthreads();
    }
    if (t == 0) sums[q] = red[0];
    __syncthreads();
  }

  if (t == 0) {
    const float invS = 1.f / (float)S;
    float aux = 0.f;
#pragma unroll
    for (int e = 0; e < E; ++e) {
      const float imp = sums[e] * invS;
      const float ld  = sums[8 + e] * invS;
      aux += imp * ld;
      tail[1 + e] = ld;
    }
    tail[0] = AUX_COEF * (float)E * aux;
    tail[9] = sums[16] * invS;
  }
}

// =====================================================================
// launch
// =====================================================================
extern "C" void kernel_launch(void* const* d_in, const int* in_sizes, int n_in,
                              void* d_out, int out_size, void* d_ws, size_t ws_size,
                              hipStream_t stream) {
  const float* x      = (const float*)d_in[0];
  const float* W_in   = (const float*)d_in[1];
  const float* b_in   = (const float*)d_in[2];
  const float* W_out  = (const float*)d_in[3];
  const float* b_out  = (const float*)d_in[4];
  const float* rw     = (const float*)d_in[5];
  const float* rb     = (const float*)d_in[6];
  float* out = (float*)d_out;

  char* ws = (char*)d_ws;
  float* final_w  = (float*)(ws + 0);                 // S*E f32 = 64 KB
  float* probs_ws = (float*)(ws + 65536);             // S*E f32 = 64 KB
  float* ent_ws   = (float*)(ws + 131072);            // S f32   =  8 KB
  bf16*  act      = (bf16*)(ws + 139264);             // S*KDIM2 bf16 ~ 90 MiB

  // 1) router: 2048 tokens, one wave each
  router_kernel<<<S / 8, 256, 0, stream>>>(x, rw, rb, final_w, probs_ws, ent_ws);

  // 2) GEMM1: M-tiles fastest-varying -> co-scheduled WGs share weight panels in L2
  gemm1_kernel<<<dim3(S / 256, FF / 32, E), 256, 0, stream>>>(
      x, W_in, b_in, final_w, act);

  // 3) GEMM2
  gemm2_kernel<<<dim3(S / 256, H / 64), 256, 0, stream>>>(
      act, W_out, b_out, final_w, out);

  // 4) aux stats -> tail of d_out
  reduce_stats_kernel<<<1, 256, 0, stream>>>(
      final_w, probs_ws, ent_ws, out + (size_t)S * H);
}
